// GADTR_11527692222687
// MI455X (gfx1250) — matrix operand order repaired and checked by reading, not compiled, not run on hardware
//
#include <hip/hip_runtime.h>
#include <hip/hip_bf16.h>

// ---------------- problem constants ----------------
#define BSZ 8
#define TT  10
#define NN  12
#define CB  512
#define OHH 24
#define OWW 42
#define HH  256
#define SS  5
#define GG  4
#define NHD 4
#define HDIM 64
#define BT  (BSZ*TT)          // 80
#define MM  (BT*NN)           // 960
#define PP  (OHH*OWW)         // 1008
#define KEMB (SS*SS*CB)       // 12800
#define BN  (BSZ*NN)          // 96
#define NCLS 9                // NC+1
#define THRESH2 0.04f

#define CDIV(a,b) (((a)+(b)-1)/(b))

typedef __attribute__((ext_vector_type(16))) _Float16 v16h;
typedef __attribute__((ext_vector_type(8)))  float    v8f;

union Frag { float4 f4[2]; v16h h; };

// ---------------- CHW -> HWC transpose (coalesced both ways via LDS) -------
__global__ void k_transpose(const float* __restrict__ in, float* __restrict__ out) {
  __shared__ float tile[16][17];
  int c0 = blockIdx.x * 16, p0 = blockIdx.y * 16, img = blockIdx.z;
  int tx = threadIdx.x, ty = threadIdx.y;
  tile[ty][tx] = in[((size_t)img * CB + (c0 + ty)) * PP + (p0 + tx)];
  __syncthreads();
  out[((size_t)img * PP + (p0 + ty)) * CB + (c0 + tx)] = tile[tx][ty];
}

// ------- fc_emb weight: f32 (H, c*25+p) -> f16 (H, p*512+c) permuted -------
__global__ void k_conv_wfc(const float* __restrict__ w, _Float16* __restrict__ out) {
  int idx = blockIdx.x * blockDim.x + threadIdx.x;
  if (idx >= HH * KEMB) return;
  int h = idx / KEMB, r = idx % KEMB;
  int p = r / CB, c = r % CB;
  out[idx] = (_Float16)w[(size_t)h * KEMB + c * (SS * SS) + p];
}

__global__ void k_f32_to_f16(const float* __restrict__ in, _Float16* __restrict__ out, int n) {
  int i = blockIdx.x * blockDim.x + threadIdx.x;
  if (i < n) out[i] = (_Float16)in[i];
}

// ---------------- ROI align (HWC feature, channel-coalesced) ----------------
__global__ void k_roi_align(const float* __restrict__ feat, const float* __restrict__ boxes,
                            _Float16* __restrict__ crops) {
  int m = blockIdx.x, t = threadIdx.x;
  int img = m / NN;
  float cx = boxes[m*4+0], cy = boxes[m*4+1], bw = boxes[m*4+2], bh = boxes[m*4+3];
  float x1 = (cx - bw*0.5f) * OWW, y1 = (cy - bh*0.5f) * OHH;
  float x2 = (cx + bw*0.5f) * OWW, y2 = (cy + bh*0.5f) * OHH;
  float sw = (x2 - x1) / (float)SS, sh = (y2 - y1) / (float)SS;
  const float* base = feat + (size_t)img * PP * CB;
  for (int p = 0; p < SS*SS; ++p) {
    int iy = p / SS, ix = p % SS;
    float px = x1 + ((float)ix + 0.5f) * sw - 0.5f;
    float py = y1 + ((float)iy + 0.5f) * sh - 0.5f;
    bool valid = (px >= -1.f) && (px <= (float)OWW) && (py >= -1.f) && (py <= (float)OHH);
    float pxc = fminf(fmaxf(px, 0.f), (float)(OWW-1));
    float pyc = fminf(fmaxf(py, 0.f), (float)(OHH-1));
    float x0f = floorf(pxc), y0f = floorf(pyc);
    float lx = pxc - x0f, ly = pyc - y0f;
    int x0 = (int)x0f, y0 = (int)y0f;
    int x1i = (x0+1 < OWW-1) ? x0+1 : OWW-1;
    int y1i = (y0+1 < OHH-1) ? y0+1 : OHH-1;
    float w00 = (1.f-ly)*(1.f-lx), w01 = (1.f-ly)*lx, w10 = ly*(1.f-lx), w11 = ly*lx;
    const float* p00 = base + (size_t)(y0 *OWW + x0 )*CB;
    const float* p01 = base + (size_t)(y0 *OWW + x1i)*CB;
    const float* p10 = base + (size_t)(y1i*OWW + x0 )*CB;
    const float* p11 = base + (size_t)(y1i*OWW + x1i)*CB;
    _Float16* outp = crops + (size_t)m * KEMB + p * CB;
    for (int c = t; c < CB; c += 256) {
      float v = w00*p00[c] + w01*p01[c] + w10*p10[c] + w11*p11[c];
      outp[c] = (_Float16)(valid ? v : 0.f);
    }
  }
}

// ---------------- generic WMMA GEMM: C = act(A(f16,MxK) @ B(f16,NxK)^T + bias) + resid
// 4 waves/block; each wave owns a 32x32 macro-tile = 2x2 WMMA tiles
// (register blocking: 8 b128 loads feed 4 WMMAs -> 2 loads/WMMA).
// A/B fragments use contiguous k order: lane half hi holds k = 16*hi + 0..15. M % 32 == 0, N % 128 == 0 per grid.
__global__ void k_wmma_gemm(const _Float16* __restrict__ A, const _Float16* __restrict__ B,
                            const float* __restrict__ bias, const float* __restrict__ resid,
                            float* __restrict__ C, int Mdim, int Ndim, int Kdim, int relu) {
  int lane  = threadIdx.x & 31;
  int wave  = threadIdx.x >> 5;
  int l15   = lane & 15;
  int hi    = lane >> 4;                        // 0: lanes 0-15, 1: lanes 16-31
  int mbase = blockIdx.x * 32;                  // 2 M-tiles
  int nbase = (blockIdx.y * 4 + wave) * 32;     // 2 N-tiles
  const _Float16* arow0 = A + (size_t)(mbase      + l15) * Kdim + hi*16;
  const _Float16* arow1 = A + (size_t)(mbase + 16 + l15) * Kdim + hi*16;
  const _Float16* brow0 = B + (size_t)(nbase      + l15) * Kdim + hi*16;
  const _Float16* brow1 = B + (size_t)(nbase + 16 + l15) * Kdim + hi*16;
  v8f acc00 = {}, acc01 = {}, acc10 = {}, acc11 = {};
  for (int k0 = 0; k0 < Kdim; k0 += 32) {
    // prefetch streaming A ~1KB ahead (global_prefetch_b8); B stays L2-resident
    __builtin_prefetch(arow0 + k0 + 512, 0, 1);
    __builtin_prefetch(arow1 + k0 + 512, 0, 1);
    Frag a0, a1, b0, b1;
    a0.f4[0] = *(const float4*)(arow0 + k0);       // K = base+0..7
    a0.f4[1] = *(const float4*)(arow0 + k0 + 8);   // K = base+8..15
    a1.f4[0] = *(const float4*)(arow1 + k0);
    a1.f4[1] = *(const float4*)(arow1 + k0 + 8);
    b0.f4[0] = *(const float4*)(brow0 + k0);       // K = base+0..7
    b0.f4[1] = *(const float4*)(brow0 + k0 + 8);   // K = base+8..15
    b1.f4[0] = *(const float4*)(brow1 + k0);
    b1.f4[1] = *(const float4*)(brow1 + k0 + 8);
    acc00 = __builtin_amdgcn_wmma_f32_16x16x32_f16(false, a0.h, false, b0.h, (short)0, acc00, false, false);
    acc01 = __builtin_amdgcn_wmma_f32_16x16x32_f16(false, a0.h, false, b1.h, (short)0, acc01, false, false);
    acc10 = __builtin_amdgcn_wmma_f32_16x16x32_f16(false, a1.h, false, b0.h, (short)0, acc10, false, false);
    acc11 = __builtin_amdgcn_wmma_f32_16x16x32_f16(false, a1.h, false, b1.h, (short)0, acc11, false, false);
  }
  int n0 = nbase + l15, n1 = n0 + 16;
  float bv0 = bias ? bias[n0] : 0.f;
  float bv1 = bias ? bias[n1] : 0.f;
  #pragma unroll
  for (int r = 0; r < 8; ++r) {
    int mA = mbase + r + hi*8;
    int mB = mA + 16;
    float v;
    v = acc00[r] + bv0; if (relu) v = fmaxf(v, 0.f);
    if (resid) v += resid[(size_t)mA*Ndim + n0];  C[(size_t)mA*Ndim + n0] = v;
    v = acc01[r] + bv1; if (relu) v = fmaxf(v, 0.f);
    if (resid) v += resid[(size_t)mA*Ndim + n1];  C[(size_t)mA*Ndim + n1] = v;
    v = acc10[r] + bv0; if (relu) v = fmaxf(v, 0.f);
    if (resid) v += resid[(size_t)mB*Ndim + n0];  C[(size_t)mB*Ndim + n0] = v;
    v = acc11[r] + bv1; if (relu) v = fmaxf(v, 0.f);
    if (resid) v += resid[(size_t)mB*Ndim + n1];  C[(size_t)mB*Ndim + n1] = v;
  }
}

// ---------------- box MLP, added into X (= relu(fc_emb) from GEMM) ----------
__global__ void k_box_mlp(const float* __restrict__ boxes,
                          const float* __restrict__ w1, const float* __restrict__ b1,
                          const float* __restrict__ w2, const float* __restrict__ b2,
                          const float* __restrict__ w3, const float* __restrict__ b3,
                          float* __restrict__ X) {
  __shared__ float h1[HH]; __shared__ float h2[HH];
  int m = blockIdx.x, t = threadIdx.x;
  float bx0 = boxes[m*4+0], bx1 = boxes[m*4+1], bx2 = boxes[m*4+2], bx3 = boxes[m*4+3];
  float s = b1[t] + w1[t*4+0]*bx0 + w1[t*4+1]*bx1 + w1[t*4+2]*bx2 + w1[t*4+3]*bx3;
  h1[t] = fmaxf(s, 0.f); __syncthreads();
  s = b2[t]; { const float* wr = w2 + (size_t)t*HH; for (int j = 0; j < HH; ++j) s += wr[j]*h1[j]; }
  h2[t] = fmaxf(s, 0.f); __syncthreads();
  s = b3[t]; { const float* wr = w3 + (size_t)t*HH; for (int j = 0; j < HH; ++j) s += wr[j]*h2[j]; }
  X[(size_t)m*HH + t] += s;
}

// ---------------- spatial (actor) attention, one block per bt ---------------
__global__ void k_fg_attn(const float* __restrict__ Q, const float* __restrict__ Kk,
                          const float* __restrict__ V, const float* __restrict__ boxes,
                          const int* __restrict__ dummy, float* __restrict__ ctx) {
  __shared__ float sc[NN*NN];
  int bt = blockIdx.x, t = threadIdx.x;
  int b = bt / TT;
  if (t < NN*NN) {
    int i = t / NN, j = t % NN;
    int mi = bt*NN + i, mj = bt*NN + j;
    const float* qi = Q + (size_t)mi*HH; const float* kj = Kk + (size_t)mj*HH;
    float s = 0.f;
    for (int d = 0; d < HH; ++d) s += qi[d]*kj[d];
    s *= (1.f/16.f);                                   // 1/sqrt(256)
    float dx = boxes[mi*4+0]-boxes[mj*4+0], dy = boxes[mi*4+1]-boxes[mj*4+1];
    int di = dummy[b*NN+i] != 0, dj = dummy[b*NN+j] != 0;
    bool mask = (dx*dx + dy*dy > THRESH2) || (di != dj);
    sc[t] = mask ? -1e9f : s;
  }
  __syncthreads();
  if (t < NN) {
    float mx = -1e30f;
    for (int j = 0; j < NN; ++j) mx = fmaxf(mx, sc[t*NN+j]);
    float sum = 0.f;
    for (int j = 0; j < NN; ++j) { float e = __expf(sc[t*NN+j]-mx); sc[t*NN+j] = e; sum += e; }
    float inv = 1.f/sum;
    for (int j = 0; j < NN; ++j) sc[t*NN+j] *= inv;
  }
  __syncthreads();
  for (int idx = t; idx < NN*HH; idx += 256) {
    int i = idx >> 8, d = idx & 255;
    float acc = 0.f;
    for (int j = 0; j < NN; ++j) acc += sc[i*NN+j] * V[(size_t)(bt*NN+j)*HH + d];
    ctx[(size_t)(bt*NN+i)*HH + d] = acc;
  }
}

// ------- (bt,n,H) -> (bn,t,H) reorder; also qk = ti + time_pos --------------
__global__ void k_reshape_ti(const float* __restrict__ gout, const float* __restrict__ tpos,
                             float* __restrict__ ti, float* __restrict__ qk) {
  int idx = blockIdx.x * blockDim.x + threadIdx.x;
  if (idx >= MM*HH) return;
  int d = idx & 255; int row = idx >> 8;
  int tt = row % TT; int bn = row / TT; int nn = bn % NN; int b = bn / NN;
  float v = gout[(size_t)((b*TT + tt)*NN + nn)*HH + d];
  ti[idx] = v;
  qk[idx] = v + tpos[tt*HH + d];
}

// ---------------- temporal multi-head attention, one block per bn -----------
__global__ void k_te_attn(const float* __restrict__ TQ, const float* __restrict__ TK,
                          const float* __restrict__ TV, float* __restrict__ TC) {
  __shared__ float sc[NHD*TT*TT];
  int bn = blockIdx.x, t = threadIdx.x;               // blockDim = 512
  if (t < NHD*TT*TT) {
    int h = t / (TT*TT), r = (t / TT) % TT, c = t % TT;
    const float* q = TQ + (size_t)(bn*TT + r)*HH + h*HDIM;
    const float* k = TK + (size_t)(bn*TT + c)*HH + h*HDIM;
    float s = 0.f;
    for (int d = 0; d < HDIM; ++d) s += q[d]*k[d];
    sc[t] = s * 0.125f;                                // 1/sqrt(64)
  }
  __syncthreads();
  if (t < NHD*TT) {
    float* row = sc + (t/TT)*TT*TT + (t%TT)*TT;
    float mx = -1e30f;
    for (int c = 0; c < TT; ++c) mx = fmaxf(mx, row[c]);
    float sum = 0.f;
    for (int c = 0; c < TT; ++c) { float e = __expf(row[c]-mx); row[c] = e; sum += e; }
    float inv = 1.f/sum;
    for (int c = 0; c < TT; ++c) row[c] *= inv;
  }
  __syncthreads();
  for (int idx = t; idx < TT*HH; idx += 512) {
    int r = idx >> 8, d = idx & 255; int h = d >> 6;
    const float* arow = sc + h*TT*TT + r*TT;
    float acc = 0.f;
    for (int c = 0; c < TT; ++c) acc += arow[c] * TV[(size_t)(bn*TT+c)*HH + d];
    TC[(size_t)(bn*TT+r)*HH + d] = acc;
  }
}

// ---------------- LayerNorm per row (H=256 = blockDim) ----------------------
__global__ void k_layernorm(const float* __restrict__ in, const float* __restrict__ g,
                            const float* __restrict__ b, float* __restrict__ out) {
  __shared__ float red[HH];
  int row = blockIdx.x, t = threadIdx.x;
  float v = in[(size_t)row*HH + t];
  red[t] = v; __syncthreads();
  for (int s = 128; s > 0; s >>= 1) { if (t < s) red[t] += red[t+s]; __syncthreads(); }
  float mu = red[0] * (1.f/HH); __syncthreads();
  float d = v - mu; red[t] = d*d; __syncthreads();
  for (int s = 128; s > 0; s >>= 1) { if (t < s) red[t] += red[t+s]; __syncthreads(); }
  float var = red[0] * (1.f/HH);
  out[(size_t)row*HH + t] = d * rsqrtf(var + 1e-5f) * g[t] + b[t];
}

// ---------------- attention pooling + l2norm(inst), one block per bn --------
__global__ void k_pool(const float* __restrict__ tout, const float* __restrict__ pw,
                       const float* __restrict__ pb, float* __restrict__ aclip,
                       float* __restrict__ inst_ws, float* __restrict__ inst_out) {
  __shared__ float logits[TT];
  __shared__ float red[HH];
  int bn = blockIdx.x, t = threadIdx.x;
  if (t < TT) {
    float s = pb[0];
    const float* row = tout + (size_t)(bn*TT + t)*HH;
    for (int d = 0; d < HH; ++d) s += pw[d]*row[d];
    logits[t] = s;
  }
  __syncthreads();
  if (t == 0) {
    float mx = -1e30f;
    for (int i = 0; i < TT; ++i) mx = fmaxf(mx, logits[i]);
    float sum = 0.f;
    for (int i = 0; i < TT; ++i) { logits[i] = __expf(logits[i]-mx); sum += logits[i]; }
    float inv = 1.f/sum;
    for (int i = 0; i < TT; ++i) logits[i] *= inv;
  }
  __syncthreads();
  float acc = 0.f;
  for (int tt = 0; tt < TT; ++tt) acc += logits[tt] * tout[(size_t)(bn*TT+tt)*HH + t];
  aclip[(size_t)bn*HH + t] = acc;
  red[t] = acc*acc; __syncthreads();
  for (int s = 128; s > 0; s >>= 1) { if (t < s) red[t] += red[t+s]; __syncthreads(); }
  float iv = acc / fmaxf(sqrtf(red[0]), 1e-12f);
  inst_ws[(size_t)bn*HH + t] = iv;
  inst_out[(size_t)bn*HH + t] = iv;
}

// ---------------- group attention, one block per batch ----------------------
__global__ void k_group_attn(const float* __restrict__ gq, const float* __restrict__ aclip,
                             float* __restrict__ grr) {
  __shared__ float sc[GG*NN];
  int b = blockIdx.x, t = threadIdx.x;
  if (t < GG*NN) {
    int g = t / NN, n = t % NN;
    const float* qg = gq + (size_t)g*HH;
    const float* an = aclip + (size_t)(b*NN+n)*HH;
    float s = 0.f;
    for (int d = 0; d < HH; ++d) s += qg[d]*an[d];
    sc[t] = s * (1.f/16.f);
  }
  __syncthreads();
  if (t < GG) {
    float* row = sc + t*NN;
    float mx = -1e30f;
    for (int n = 0; n < NN; ++n) mx = fmaxf(mx, row[n]);
    float sum = 0.f;
    for (int n = 0; n < NN; ++n) { float e = __expf(row[n]-mx); row[n] = e; sum += e; }
    float inv = 1.f/sum;
    for (int n = 0; n < NN; ++n) row[n] *= inv;
  }
  __syncthreads();
  for (int idx = t; idx < GG*HH; idx += 256) {
    int g = idx >> 8, d = idx & 255;
    float acc = 0.f;
    for (int n = 0; n < NN; ++n) acc += sc[g*NN+n] * aclip[(size_t)(b*NN+n)*HH + d];
    grr[(size_t)(b*GG+g)*HH + d] = acc;
  }
}

__global__ void k_l2norm(const float* __restrict__ in, float* __restrict__ out) {
  __shared__ float red[HH];
  int row = blockIdx.x, t = threadIdx.x;
  float v = in[(size_t)row*HH + t];
  red[t] = v*v; __syncthreads();
  for (int s = 128; s > 0; s >>= 1) { if (t < s) red[t] += red[t+s]; __syncthreads(); }
  out[(size_t)row*HH + t] = v / fmaxf(sqrtf(red[0]), 1e-12f);
}

__global__ void k_linear_small(const float* __restrict__ A, const float* __restrict__ W,
                               const float* __restrict__ b, float* __restrict__ out,
                               int rows, int cols) {
  int idx = blockIdx.x * blockDim.x + threadIdx.x;
  if (idx >= rows*cols) return;
  int r = idx / cols, c = idx % cols;
  const float* ar = A + (size_t)r*HH;
  const float* wr = W + (size_t)c*HH;
  float s = b[c];
  for (int d = 0; d < HH; ++d) s += ar[d]*wr[d];
  out[idx] = s;
}

__global__ void k_membership(const float* __restrict__ GE, const float* __restrict__ AE,
                             float* __restrict__ out) {
  __shared__ float sc[GG*NN];
  int b = blockIdx.x, t = threadIdx.x;
  if (t < GG*NN) {
    int g = t / NN, n = t % NN;
    const float* ge = GE + (size_t)(b*GG+g)*HH;
    const float* ae = AE + (size_t)(b*NN+n)*HH;
    float s = 0.f;
    for (int d = 0; d < HH; ++d) s += ge[d]*ae[d];
    sc[t] = s;
  }
  __syncthreads();
  if (t < NN) {
    float mx = -1e30f;
    for (int g = 0; g < GG; ++g) mx = fmaxf(mx, sc[g*NN+t]);
    float e[GG]; float sum = 0.f;
    for (int g = 0; g < GG; ++g) { e[g] = __expf(sc[g*NN+t]-mx); sum += e[g]; }
    float inv = 1.f/sum;
    for (int g = 0; g < GG; ++g) out[b*GG*NN + g*NN + t] = e[g]*inv;
  }
}

// ===================== host side =====================
static inline size_t alignup(size_t x) { return (x + 255) & ~(size_t)255; }

extern "C" void kernel_launch(void* const* d_in, const int* in_sizes, int n_in,
                              void* d_out, int out_size, void* d_ws, size_t ws_size,
                              hipStream_t stream) {
  (void)in_sizes; (void)n_in; (void)out_size; (void)ws_size;
  const float* features = (const float*)d_in[0];
  const float* boxes    = (const float*)d_in[1];
  const int*   dummy    = (const int*)d_in[2];
  const float* fc_emb_w = (const float*)d_in[3];
  const float* fc_emb_b = (const float*)d_in[4];
  const float* mlp_w1   = (const float*)d_in[5];
  const float* mlp_b1   = (const float*)d_in[6];
  const float* mlp_w2   = (const float*)d_in[7];
  const float* mlp_b2   = (const float*)d_in[8];
  const float* mlp_w3   = (const float*)d_in[9];
  const float* mlp_b3   = (const float*)d_in[10];
  const float* fg_q_w = (const float*)d_in[11]; const float* fg_q_b = (const float*)d_in[12];
  const float* fg_k_w = (const float*)d_in[13]; const float* fg_k_b = (const float*)d_in[14];
  const float* fg_v_w = (const float*)d_in[15]; const float* fg_v_b = (const float*)d_in[16];
  const float* fg_o_w = (const float*)d_in[17]; const float* fg_o_b = (const float*)d_in[18];
  const float* te_q_w = (const float*)d_in[19]; const float* te_q_b = (const float*)d_in[20];
  const float* te_k_w = (const float*)d_in[21]; const float* te_k_b = (const float*)d_in[22];
  const float* te_v_w = (const float*)d_in[23]; const float* te_v_b = (const float*)d_in[24];
  const float* te_o_w = (const float*)d_in[25]; const float* te_o_b = (const float*)d_in[26];
  const float* am_w   = (const float*)d_in[27]; const float* am_b   = (const float*)d_in[28];
  const float* gm_w   = (const float*)d_in[29]; const float* gm_b   = (const float*)d_in[30];
  const float* te_ln_g = (const float*)d_in[31];
  const float* te_ln_b = (const float*)d_in[32];
  const float* time_pos = (const float*)d_in[33];
  const float* group_query = (const float*)d_in[34];
  const float* class_w = (const float*)d_in[35]; const float* class_b = (const float*)d_in[36];
  const float* group_w = (const float*)d_in[37]; const float* group_b = (const float*)d_in[38];
  const float* pool_w  = (const float*)d_in[39]; const float* pool_b  = (const float*)d_in[40];

  char* ws = (char*)d_ws;
  size_t off = 0;
  auto carve = [&](size_t bytes) { char* p = ws + off; off = alignup(off + bytes); return p; };
  float*    FEAT   = (float*)   carve((size_t)BT*PP*CB*4);       // 165.2 MB
  _Float16* CROPS  = (_Float16*)carve((size_t)MM*KEMB*2);        // 24.6 MB
  _Float16* WFC16  = (_Float16*)carve((size_t)HH*KEMB*2);        // 6.55 MB
  _Float16* ACT16  = (_Float16*)carve((size_t)MM*HH*2);
  _Float16* W16    = (_Float16*)carve((size_t)HH*HH*2);
  const size_t MH = (size_t)MM*HH*4;
  float* X    = (float*)carve(MH);  float* Qb   = (float*)carve(MH);
  float* Kb   = (float*)carve(MH);  float* Vb   = (float*)carve(MH);
  float* CTX  = (float*)carve(MH);  float* GOUT = (float*)carve(MH);
  float* TI   = (float*)carve(MH);  float* QKB  = (float*)carve(MH);
  float* TQ   = (float*)carve(MH);  float* TK   = (float*)carve(MH);
  float* TV   = (float*)carve(MH);  float* TCTX = (float*)carve(MH);
  float* PRELN= (float*)carve(MH);  float* TOUT = (float*)carve(MH);
  float* ACLIP = (float*)carve((size_t)BN*HH*4);
  float* INST  = (float*)carve((size_t)BN*HH*4);
  float* GRR   = (float*)carve((size_t)BSZ*GG*HH*4);
  float* GRN   = (float*)carve((size_t)BSZ*GG*HH*4);
  float* AE    = (float*)carve((size_t)BN*HH*4);
  float* GE    = (float*)carve((size_t)BSZ*GG*HH*4);

  float* out_actions    = (float*)d_out;                 // (8,12,9)   = 864
  float* out_activities = out_actions + BSZ*NN*NCLS;     // (8,4,9)    = 288
  float* out_membership = out_activities + BSZ*GG*NCLS;  // (8,4,12)   = 384
  float* out_inst       = out_membership + BSZ*GG*NN;    // (8,12,256) = 24576

  // 1) CHW -> HWC
  k_transpose<<<dim3(CB/16, PP/16, BT), dim3(16,16), 0, stream>>>(features, FEAT);
  // 2) fc_emb weight -> f16, K permuted to p*512+c
  k_conv_wfc<<<CDIV(HH*KEMB,256), 256, 0, stream>>>(fc_emb_w, WFC16);
  // 3) ROI align -> f16 crops (960 x 12800)
  k_roi_align<<<MM, 256, 0, stream>>>(FEAT, boxes, CROPS);
  // 4) big WMMA GEMM: X = relu(crops @ Wfc^T + b)   (32x32 macro-tiles)
  k_wmma_gemm<<<dim3(MM/32, HH/128), 128, 0, stream>>>(CROPS, WFC16, fc_emb_b, nullptr,
                                                       X, MM, HH, KEMB, 1);
  // 5) X += boxMLP(boxes)
  k_box_mlp<<<MM, HH, 0, stream>>>(boxes, mlp_w1, mlp_b1, mlp_w2, mlp_b2, mlp_w3, mlp_b3, X);

  auto gemm256 = [&](const float* act, const float* w, const float* b, const float* res,
                     float* out, int Mrows, int relu) {
    k_f32_to_f16<<<CDIV(Mrows*HH,256), 256, 0, stream>>>(act, ACT16, Mrows*HH);
    k_f32_to_f16<<<CDIV(HH*HH,256), 256, 0, stream>>>(w, W16, HH*HH);
    k_wmma_gemm<<<dim3(Mrows/32, HH/128), 128, 0, stream>>>(ACT16, W16, b, res,
                                                            out, Mrows, HH, HH, relu);
  };

  // 6) spatial attention projections + attention + output proj (residual X)
  k_f32_to_f16<<<CDIV(MM*HH,256), 256, 0, stream>>>(X, ACT16, MM*HH);
  k_f32_to_f16<<<CDIV(HH*HH,256), 256, 0, stream>>>(fg_q_w, W16, HH*HH);
  k_wmma_gemm<<<dim3(MM/32, HH/128), 128, 0, stream>>>(ACT16, W16, fg_q_b, nullptr, Qb, MM, HH, HH, 0);
  k_f32_to_f16<<<CDIV(HH*HH,256), 256, 0, stream>>>(fg_k_w, W16, HH*HH);
  k_wmma_gemm<<<dim3(MM/32, HH/128), 128, 0, stream>>>(ACT16, W16, fg_k_b, nullptr, Kb, MM, HH, HH, 0);
  k_f32_to_f16<<<CDIV(HH*HH,256), 256, 0, stream>>>(fg_v_w, W16, HH*HH);
  k_wmma_gemm<<<dim3(MM/32, HH/128), 128, 0, stream>>>(ACT16, W16, fg_v_b, nullptr, Vb, MM, HH, HH, 0);
  k_fg_attn<<<BT, 256, 0, stream>>>(Qb, Kb, Vb, boxes, dummy, CTX);
  gemm256(CTX, fg_o_w, fg_o_b, X, GOUT, MM, 0);

  // 7) temporal encoder
  k_reshape_ti<<<CDIV(MM*HH,256), 256, 0, stream>>>(GOUT, time_pos, TI, QKB);
  k_f32_to_f16<<<CDIV(MM*HH,256), 256, 0, stream>>>(QKB, ACT16, MM*HH);
  k_f32_to_f16<<<CDIV(HH*HH,256), 256, 0, stream>>>(te_q_w, W16, HH*HH);
  k_wmma_gemm<<<dim3(MM/32, HH/128), 128, 0, stream>>>(ACT16, W16, te_q_b, nullptr, TQ, MM, HH, HH, 0);
  k_f32_to_f16<<<CDIV(HH*HH,256), 256, 0, stream>>>(te_k_w, W16, HH*HH);
  k_wmma_gemm<<<dim3(MM/32, HH/128), 128, 0, stream>>>(ACT16, W16, te_k_b, nullptr, TK, MM, HH, HH, 0);
  gemm256(TI, te_v_w, te_v_b, nullptr, TV, MM, 0);
  k_te_attn<<<BN, 512, 0, stream>>>(TQ, TK, TV, TCTX);
  gemm256(TCTX, te_o_w, te_o_b, TI, PRELN, MM, 0);
  k_layernorm<<<MM, HH, 0, stream>>>(PRELN, te_ln_g, te_ln_b, TOUT);

  // 8) pooling + inst (l2norm), group attention, heads
  k_pool<<<BN, HH, 0, stream>>>(TOUT, pool_w, pool_b, ACLIP, INST, out_inst);
  k_group_attn<<<BSZ, 256, 0, stream>>>(group_query, ACLIP, GRR);
  k_l2norm<<<BSZ*GG, HH, 0, stream>>>(GRR, GRN);
  k_linear_small<<<CDIV(BN*NCLS,256), 256, 0, stream>>>(ACLIP, class_w, class_b,
                                                        out_actions, BN, NCLS);
  k_linear_small<<<CDIV(BSZ*GG*NCLS,256), 256, 0, stream>>>(GRR, group_w, group_b,
                                                            out_activities, BSZ*GG, NCLS);
  // ae = am(inst), ge = gm(grn) via WMMA (M=96 and M=32, both multiples of 32)
  gemm256(INST, am_w, am_b, nullptr, AE, BN, 0);
  gemm256(GRN,  gm_w, gm_b, nullptr, GE, BSZ*GG, 0);
  k_membership<<<BSZ, 256, 0, stream>>>(GE, AE, out_membership);
}
